// CausalSelfAttention_39745627357279
// MI455X (gfx1250) — matrix operand order repaired and checked by reading, not compiled, not run on hardware
//
#include <hip/hip_runtime.h>
#include <cstdint>

typedef __attribute__((ext_vector_type(16))) __bf16 v16bf;
typedef __attribute__((ext_vector_type(8)))  float  v8f;

struct alignas(16) B128 { unsigned int x[4]; };
union Frag { v16bf v; unsigned short u[16]; B128 q4[2]; };

__device__ __forceinline__ unsigned short f2bf(float f) {
    union { float f; unsigned int u; } cv; cv.f = f;
    unsigned int u = cv.u;
    unsigned int rb = 0x7FFFu + ((u >> 16) & 1u);   // round-to-nearest-even
    return (unsigned short)((u + rb) >> 16);
}

__device__ __forceinline__ v8f wmma_bf16(const Frag& a, const Frag& b, v8f c) {
    return __builtin_amdgcn_wmma_f32_16x16x32_bf16(false, a.v, false, b.v,
                                                   (short)0, c, false, false);
}

// ---------------------------------------------------------------- cast f32->bf16
__global__ void cast_bf16(const float* __restrict__ in, unsigned short* __restrict__ out, int n) {
    int i = blockIdx.x * blockDim.x + threadIdx.x;
    if (i < n) out[i] = f2bf(in[i]);
}

// ---------------------------------------------------------------- NT GEMM via WMMA
// C[M,N] fp32 = A[M,K] bf16 (row major) * B[N,K] bf16 (row major, used transposed)
// grid: (M/64, N/256), block: 256 (8 waves as 2x4; each wave: 32x64 macro-tile).
// K-loop unrolled x2 with two named register buffers -> no rotation movs,
// unconditional loads, software-pipelined. Requires K % 64 == 0, K >= 128.
__global__ __launch_bounds__(256) void gemm_bf16_nt(
        const unsigned short* __restrict__ A, const unsigned short* __restrict__ Bw,
        float* __restrict__ C, int M, int N, int K) {
    int lane = threadIdx.x & 31;
    int wave = threadIdx.x >> 5;
    int h = lane >> 4, n16 = lane & 15;
    int wm = wave & 1, wn = wave >> 1;
    int m0 = blockIdx.x * 64 + wm * 32;        // rows m0..m0+31 (two 16-row tiles)
    int nb = blockIdx.y * 256 + wn * 64;       // cols nb..nb+63 (four 16-col tiles)

    const unsigned short* ar0 = A + ((size_t)m0 + n16) * K;
    const unsigned short* ar1 = A + ((size_t)m0 + 16 + n16) * K;
    const unsigned short* br0 = Bw + ((size_t)nb + n16) * K;
    const unsigned short* br1 = Bw + ((size_t)nb + 16 + n16) * K;
    const unsigned short* br2 = Bw + ((size_t)nb + 32 + n16) * K;
    const unsigned short* br3 = Bw + ((size_t)nb + 48 + n16) * K;

    v8f acc[8];                                // [mi*4 + cb]
    for (int i = 0; i < 8; ++i)
        for (int r = 0; r < 8; ++r) acc[i][r] = 0.0f;

    auto loadA = [&](Frag& f, const unsigned short* row, int kk) {
        f.q4[0] = *(const B128*)(row + kk + 16 * h);       // K = 16h..16h+7
        f.q4[1] = *(const B128*)(row + kk + 16 * h + 8);   // K = 16h+8..16h+15
    };
    auto loadB = [&](Frag& f, const unsigned short* row, int kk) {
        f.q4[0] = *(const B128*)(row + kk + 16 * h);       // K = 16h..16h+15
        f.q4[1] = *(const B128*)(row + kk + 16 * h + 8);
    };

    Frag xa0, xa1, xb0, xb1, xb2, xb3;         // buffer X
    Frag ya0, ya1, yb0, yb1, yb2, yb3;         // buffer Y

    auto loadX = [&](int kk) {
        loadA(xa0, ar0, kk); loadA(xa1, ar1, kk);
        loadB(xb0, br0, kk); loadB(xb1, br1, kk);
        loadB(xb2, br2, kk); loadB(xb3, br3, kk);
    };
    auto loadY = [&](int kk) {
        loadA(ya0, ar0, kk); loadA(ya1, ar1, kk);
        loadB(yb0, br0, kk); loadB(yb1, br1, kk);
        loadB(yb2, br2, kk); loadB(yb3, br3, kk);
    };
    auto computeX = [&]() {
        acc[0] = wmma_bf16(xa0, xb0, acc[0]);
        acc[4] = wmma_bf16(xa1, xb0, acc[4]);
        acc[1] = wmma_bf16(xa0, xb1, acc[1]);
        acc[5] = wmma_bf16(xa1, xb1, acc[5]);
        acc[2] = wmma_bf16(xa0, xb2, acc[2]);
        acc[6] = wmma_bf16(xa1, xb2, acc[6]);
        acc[3] = wmma_bf16(xa0, xb3, acc[3]);
        acc[7] = wmma_bf16(xa1, xb3, acc[7]);
    };
    auto computeY = [&]() {
        acc[0] = wmma_bf16(ya0, yb0, acc[0]);
        acc[4] = wmma_bf16(ya1, yb0, acc[4]);
        acc[1] = wmma_bf16(ya0, yb1, acc[1]);
        acc[5] = wmma_bf16(ya1, yb1, acc[5]);
        acc[2] = wmma_bf16(ya0, yb2, acc[2]);
        acc[6] = wmma_bf16(ya1, yb2, acc[6]);
        acc[3] = wmma_bf16(ya0, yb3, acc[3]);
        acc[7] = wmma_bf16(ya1, yb3, acc[7]);
    };

    loadX(0);
    int kk = 0;
    for (; kk < K - 64; kk += 64) {
        loadY(kk + 32);
        __builtin_prefetch((const void*)(ar0 + kk + 64), 0, 1);
        __builtin_prefetch((const void*)(br0 + kk + 64), 0, 1);
        computeX();
        loadX(kk + 64);
        computeY();
    }
    loadY(kk + 32);                            // kk == K-64 here
    computeX();
    computeY();

    for (int mi = 0; mi < 2; ++mi)
        for (int cb = 0; cb < 4; ++cb) {
            float* crow = C + ((size_t)m0 + mi * 16) * N + nb + cb * 16 + n16;
            for (int r = 0; r < 8; ++r)
                crow[(size_t)(r + 8 * h) * N] = acc[mi * 4 + cb][r];
        }
}

// ---------------------------------------------------------------- RMSnorm + RoPE (+gain)
// raw: [B*T, Hn*64] fp32 (GEMM output). out: [B, Hn, T, 64] bf16.
// grid: (B*T, Hn), block: 64
__global__ void finalize_qk(const float* __restrict__ raw, const float* __restrict__ gain,
                            unsigned short* __restrict__ outbf, int Hn) {
    const int T = 2048;
    int d  = threadIdx.x;                 // 0..63
    int hh = blockIdx.y;
    int bt = blockIdx.x;
    int t  = bt & (T - 1);
    int b  = bt >> 11;

    float v = raw[(size_t)bt * (Hn * 64) + hh * 64 + d];

    __shared__ float red[64];
    __shared__ float vals[64];
    red[d] = v * v;
    __syncthreads();
    for (int o = 32; o > 0; o >>= 1) { if (d < o) red[d] += red[d + o]; __syncthreads(); }
    float rn = rsqrtf(red[0] * (1.0f / 64.0f) + 1.1920929e-07f);
    vals[d] = v * rn;
    __syncthreads();

    int i = d & 31;
    float inv = __expf(-(float)i * (9.210340371976184f / 32.0f)); // 10000^(-i/32)
    float f = (float)t * inv;
    float sn, cs;
    __sincosf(f, &sn, &cs);
    float x1 = vals[i], x2 = vals[i + 32];
    float o = (d < 32) ? (x1 * cs + x2 * sn) : (-x1 * sn + x2 * cs);
    if (gain) o *= gain[hh];
    outbf[(((size_t)b * Hn + hh) * T + t) * 64 + d] = f2bf(o);
}

// ---------------------------------------------------------------- V transpose + cast
// vraw: [B*T, 256] fp32 -> vt: [B, 4, T/32, 64, 32] bf16   (V^T per 32-key chunk)
__global__ void finalize_v(const float* __restrict__ raw, unsigned short* __restrict__ vt) {
    const int T = 2048;
    int i = blockIdx.x * blockDim.x + threadIdx.x;   // < 2*4*64*64*32 = 2^20
    int kidx  = i & 31;
    int d     = (i >> 5) & 63;
    int chunk = (i >> 11) & 63;          // T/32 = 64
    int g     = (i >> 17) & 3;
    int b     = i >> 19;
    int t = chunk * 32 + kidx;
    vt[i] = f2bf(raw[((size_t)(b * T + t)) * 256 + g * 64 + d]);
}

// ---------------------------------------------------------------- flash attention, WMMA bf16
// qn [B,16,T,64], kn [B,4,T,64], vt [B,4,T/32,64,32] bf16 -> yb [B,T,1024] bf16
// grid (T/32, 16, B), block 32: one wave handles 32 query rows (two 16-row tiles
// sharing K/V fragments -> 16 WMMAs per 32-key chunk).
__global__ __launch_bounds__(32) void attn_fa_wmma(
        const unsigned short* __restrict__ qn, const unsigned short* __restrict__ kn,
        const unsigned short* __restrict__ vt, unsigned short* __restrict__ yb) {
    const int T = 2048, HD = 64, H = 16, HKV = 4;
    int lane = threadIdx.x & 31;
    int h = lane >> 4, n16 = lane & 15;
    int q32  = blockIdx.x;                 // 32-query block
    int head = blockIdx.y;
    int b    = blockIdx.z;
    int g    = head >> 2;                  // kv head (rep = 4)

    const unsigned short* qbase  = qn + (((size_t)b * H + head) * T + (size_t)q32 * 32) * HD;
    const unsigned short* kbase  = kn + ((size_t)b * HKV + g) * T * HD;
    const unsigned short* vtbase = vt + ((size_t)b * HKV + g) * (T / 32) * (HD * 32);

    // Q fragments: [tile][K-chunk], resident for the whole pass
    Frag qa[2][2];
    for (int ti = 0; ti < 2; ++ti) {
        const unsigned short* qrow = qbase + ((size_t)ti * 16 + n16) * HD;
        qa[ti][0].q4[0] = *(const B128*)(qrow + 16 * h);
        qa[ti][0].q4[1] = *(const B128*)(qrow + 16 * h + 8);
        qa[ti][1].q4[0] = *(const B128*)(qrow + 32 + 16 * h);
        qa[ti][1].q4[1] = *(const B128*)(qrow + 40 + 16 * h);
    }

    __shared__ alignas(16) unsigned short P[2][16 * 32];   // probability tiles

    float rowmax[2][8], rowsum[2][8];
    v8f O[2][4];
    for (int ti = 0; ti < 2; ++ti)
        for (int r = 0; r < 8; ++r) {
            rowmax[ti][r] = -3.0e38f; rowsum[ti][r] = 0.0f;
            O[ti][0][r] = 0.f; O[ti][1][r] = 0.f; O[ti][2][r] = 0.f; O[ti][3][r] = 0.f;
        }

    const float sc = 0.125f;               // 1/sqrt(64)
    int qmaxAll = q32 * 32 + 31;

    for (int c0 = 0; c0 <= qmaxAll; c0 += 32) {
        // ---- K fragments (shared by both query tiles): keys c0..c0+15, c0+16..c0+31
        Frag f0a, f1a, f0b, f1b;
        {
            const unsigned short* k0 = kbase + (size_t)(c0 + n16) * HD;
            const unsigned short* k1 = kbase + (size_t)(c0 + 16 + n16) * HD;
            f0a.q4[0] = *(const B128*)(k0 + 16 * h);      f0a.q4[1] = *(const B128*)(k0 + 16 * h + 8);
            f1a.q4[0] = *(const B128*)(k1 + 16 * h);      f1a.q4[1] = *(const B128*)(k1 + 16 * h + 8);
            f0b.q4[0] = *(const B128*)(k0 + 32 + 16 * h); f0b.q4[1] = *(const B128*)(k0 + 40 + 16 * h);
            f1b.q4[0] = *(const B128*)(k1 + 32 + 16 * h); f1b.q4[1] = *(const B128*)(k1 + 40 + 16 * h);
        }
        // ---- V fragments from the pre-transposed global V^T (L2-resident)
        Frag vb[4];
        {
            const unsigned short* vtc = vtbase + (size_t)(c0 >> 5) * (HD * 32);
            for (int cb = 0; cb < 4; ++cb) {
                vb[cb].q4[0] = *(const B128*)(vtc + (cb * 16 + n16) * 32 + 16 * h);
                vb[cb].q4[1] = *(const B128*)(vtc + (cb * 16 + n16) * 32 + 16 * h + 8);
            }
        }

        // ---- scores + online softmax per query tile
        for (int ti = 0; ti < 2; ++ti) {
            v8f sA, sB;
            for (int r = 0; r < 8; ++r) { sA[r] = 0.f; sB[r] = 0.f; }
            sA = wmma_bf16(qa[ti][0], f0a, sA);
            sB = wmma_bf16(qa[ti][0], f1a, sB);
            sA = wmma_bf16(qa[ti][1], f0b, sA);
            sB = wmma_bf16(qa[ti][1], f1b, sB);

            float s0[8], s1[8], cm[8];
            for (int r = 0; r < 8; ++r) {
                int m = q32 * 32 + ti * 16 + r + 8 * h;
                s0[r] = (c0 + n16      <= m) ? sA[r] * sc : -3.0e38f;
                s1[r] = (c0 + 16 + n16 <= m) ? sB[r] * sc : -3.0e38f;
                cm[r] = fmaxf(s0[r], s1[r]);
            }
            for (int msk = 1; msk < 16; msk <<= 1)
                for (int r = 0; r < 8; ++r) cm[r] = fmaxf(cm[r], __shfl_xor(cm[r], msk));

            for (int r = 0; r < 8; ++r) {
                float mn = fmaxf(rowmax[ti][r], cm[r]);
                float alpha = __expf(rowmax[ti][r] - mn);
                rowmax[ti][r] = mn;
                float p0 = __expf(s0[r] - mn);
                float p1 = __expf(s1[r] - mn);
                float ps = p0 + p1;
                for (int msk = 1; msk < 16; msk <<= 1) ps += __shfl_xor(ps, msk);
                rowsum[ti][r] = rowsum[ti][r] * alpha + ps;
                O[ti][0][r] *= alpha; O[ti][1][r] *= alpha;
                O[ti][2][r] *= alpha; O[ti][3][r] *= alpha;
                P[ti][(r + 8 * h) * 32 + n16]      = f2bf(p0);
                P[ti][(r + 8 * h) * 32 + 16 + n16] = f2bf(p1);
            }
        }
        __syncthreads();   // single-wave workgroup: barrier is a NOP; LDS deps via DScnt

        // ---- O += P(16x32) * V(32x64)
        for (int ti = 0; ti < 2; ++ti) {
            Frag pa;
            pa.q4[0] = *(const B128*)&P[ti][n16 * 32 + 16 * h];
            pa.q4[1] = *(const B128*)&P[ti][n16 * 32 + 16 * h + 8];
            for (int cb = 0; cb < 4; ++cb)
                O[ti][cb] = wmma_bf16(pa, vb[cb], O[ti][cb]);
        }
        __syncthreads();
    }

    // ---- epilogue: normalize and write y as [B,T,1024] bf16
    for (int ti = 0; ti < 2; ++ti) {
        size_t trow = (size_t)b * T + (size_t)q32 * 32 + ti * 16;
        for (int cb = 0; cb < 4; ++cb)
            for (int r = 0; r < 8; ++r) {
                float yv = O[ti][cb][r] / rowsum[ti][r];
                yb[(trow + r + 8 * h) * 1024 + head * 64 + cb * 16 + n16] = f2bf(yv);
            }
    }
}

// ----------------------------------------------------------------
extern "C" void kernel_launch(void* const* d_in, const int* in_sizes, int n_in,
                              void* d_out, int out_size, void* d_ws, size_t ws_size,
                              hipStream_t stream) {
    const float* x  = (const float*)d_in[0];
    const float* Wq = (const float*)d_in[1];
    const float* Wk = (const float*)d_in[2];
    const float* Wv = (const float*)d_in[3];
    const float* Wp = (const float*)d_in[4];
    const float* qg = (const float*)d_in[5];

    const int B = 2, T = 2048, D = 1024, H = 16, HKV = 4, HD = 64;
    const size_t BT = (size_t)B * T;            // 4096
    const size_t KVD = (size_t)HKV * HD;        // 256

    char* p = (char*)d_ws;
    auto take = [&](size_t bytes) { char* r = p; p += (bytes + 255) & ~(size_t)255; return r; };
    unsigned short* xb  = (unsigned short*)take(BT * D * 2);
    unsigned short* wqb = (unsigned short*)take((size_t)D * D * 2);
    unsigned short* wkb = (unsigned short*)take(KVD * D * 2);
    unsigned short* wvb = (unsigned short*)take(KVD * D * 2);
    unsigned short* wpb = (unsigned short*)take((size_t)D * D * 2);
    float* qraw = (float*)take(BT * D * 4);
    float* kraw = (float*)take(BT * KVD * 4);
    float* vraw = (float*)take(BT * KVD * 4);
    unsigned short* qn  = (unsigned short*)take(BT * D * 2);
    unsigned short* kn  = (unsigned short*)take(BT * KVD * 2);
    unsigned short* vtb = (unsigned short*)take(BT * KVD * 2);
    unsigned short* ybf = (unsigned short*)take(BT * D * 2);

    auto cg = [](size_t n) { return dim3((unsigned)((n + 255) / 256)); };
    cast_bf16<<<cg(BT * D),        256, 0, stream>>>(x,  xb,  (int)(BT * D));
    cast_bf16<<<cg((size_t)D * D), 256, 0, stream>>>(Wq, wqb, D * D);
    cast_bf16<<<cg(KVD * D),       256, 0, stream>>>(Wk, wkb, (int)(KVD * D));
    cast_bf16<<<cg(KVD * D),       256, 0, stream>>>(Wv, wvb, (int)(KVD * D));
    cast_bf16<<<cg((size_t)D * D), 256, 0, stream>>>(Wp, wpb, D * D);

    gemm_bf16_nt<<<dim3(BT / 64, D / 256),   256, 0, stream>>>(xb, wqb, qraw, (int)BT, D, D);
    gemm_bf16_nt<<<dim3(BT / 64, 256 / 256), 256, 0, stream>>>(xb, wkb, kraw, (int)BT, 256, D);
    gemm_bf16_nt<<<dim3(BT / 64, 256 / 256), 256, 0, stream>>>(xb, wvb, vraw, (int)BT, 256, D);

    finalize_qk<<<dim3((unsigned)BT, H),   64, 0, stream>>>(qraw, qg,      qn, H);
    finalize_qk<<<dim3((unsigned)BT, HKV), 64, 0, stream>>>(kraw, nullptr, kn, HKV);
    finalize_v<<<dim3((unsigned)(BT * KVD / 256)), 256, 0, stream>>>(vraw, vtb);

    attn_fa_wmma<<<dim3(T / 32, H, B), 32, 0, stream>>>(qn, kn, vtb, ybf);

    gemm_bf16_nt<<<dim3(BT / 64, D / 256), 256, 0, stream>>>(ybf, wpb, (float*)d_out, (int)BT, D, D);
}
